// ExpertMLPsV2_30425548324864
// MI455X (gfx1250) — matrix operand order repaired and checked by reading, not compiled, not run on hardware
//
#include <hip/hip_runtime.h>
#include <stdint.h>

// Problem constants (from reference)
#define TT 4096
#define HH 2048
#define II 1408
#define EE 8
#define KK 2

typedef __attribute__((ext_vector_type(16))) __bf16 v16bf;
typedef __attribute__((ext_vector_type(8)))  __bf16 v8bf;
typedef __attribute__((ext_vector_type(8)))  float  v8f;

union BF16x16 { v16bf v; v8bf h[2]; };

// Routed slot capacity: T*K real rows + up to 63 pad rows per expert (64-row tiles)
constexpr int SLOT_CAP = TT * KK + 64 * EE;          // 8704, multiple of 64

// Workspace layout (bytes), all 16B aligned
constexpr size_t WS_COUNTS = 0;                                   // 8 counts + 8 fill
constexpr size_t WS_OFFP   = 64;                                  // 9 ints
constexpr size_t WS_TOK    = 128;                                 // SLOT_CAP ints
constexpr size_t WS_WGT    = WS_TOK  + (size_t)SLOT_CAP * 4;      // SLOT_CAP floats
constexpr size_t WS_XB     = WS_WGT  + (size_t)SLOT_CAP * 4;      // T*H bf16
constexpr size_t WS_HBUF   = WS_XB   + (size_t)TT * HH * 2;       // SLOT_CAP*I bf16
constexpr size_t WS_WGUT   = WS_HBUF + (size_t)SLOT_CAP * II * 2; // E*2I*H bf16 (transposed)
constexpr size_t WS_WDT    = WS_WGUT + (size_t)EE * 2 * II * HH * 2; // E*H*I bf16 (transposed)

// ---------------------------------------------------------------------------
// CDNA5 async global->LDS copy (ASYNCcnt path) + WMMA helper
// ---------------------------------------------------------------------------
__device__ __forceinline__ void async_b128(void* lds_dst, const void* gsrc) {
    unsigned l = (unsigned)(size_t)lds_dst;            // flat LDS addr[31:0] = LDS offset
    asm volatile("global_load_async_to_lds_b128 %0, %1, off"
                 :: "v"(l), "v"(gsrc) : "memory");
}
__device__ __forceinline__ void wait_async0() {
#if __has_builtin(__builtin_amdgcn_s_wait_asynccnt)
    __builtin_amdgcn_s_wait_asynccnt(0);
#else
    asm volatile("s_wait_asynccnt 0x0" ::: "memory");
#endif
}
__device__ __forceinline__ v8f wmma_bf16(const BF16x16& a, const BF16x16& b, v8f c) {
    return __builtin_amdgcn_wmma_f32_16x16x32_bf16(
        false, a.v, false, b.v, (short)0, c, false, false);
}

// ---------------------------------------------------------------------------
// Routing kernels
// ---------------------------------------------------------------------------
__global__ void moe_init(float* __restrict__ out, int* __restrict__ cnt,
                         int* __restrict__ tokl) {
    int i = blockIdx.x * blockDim.x + threadIdx.x;
    if (i < TT * HH) out[i] = 0.0f;
    if (i < 16)       cnt[i] = 0;
    if (i < SLOT_CAP) tokl[i] = -1;
}

__global__ void moe_count(const int* __restrict__ eidx, int* __restrict__ cnt) {
    int i = blockIdx.x * blockDim.x + threadIdx.x;
    if (i < TT * KK) atomicAdd(&cnt[eidx[i] & 7], 1);
}

__global__ void moe_scan(const int* __restrict__ cnt, int* __restrict__ offP) {
    if (threadIdx.x == 0) {
        int off = 0;
        for (int e = 0; e < EE; ++e) {
            offP[e] = off;
            off += ((cnt[e] + 63) >> 6) << 6;   // pad each expert to 64-row tiles
        }
        offP[EE] = off;
    }
}

__global__ void moe_scatter(const int* __restrict__ eidx, const float* __restrict__ aff,
                            const int* __restrict__ offP, int* __restrict__ fill,
                            int* __restrict__ tokl, float* __restrict__ wgtl) {
    int i = blockIdx.x * blockDim.x + threadIdx.x;
    if (i < TT * KK) {
        int e = eidx[i] & 7;
        int t = i / KK;
        int pos  = atomicAdd(&fill[e], 1);
        int slot = offP[e] + pos;
        tokl[slot] = t;
        wgtl[slot] = aff[t * EE + e];
    }
}

// ---------------------------------------------------------------------------
// One-time precision/layout conversion
// ---------------------------------------------------------------------------
__global__ __launch_bounds__(256) void cvt_x(const float* __restrict__ X,
                                             __bf16* __restrict__ Xb) {
    size_t i = (size_t)(blockIdx.x * 256 + threadIdx.x) * 4;
    float4 v = *(const float4*)(X + i);
    union { __bf16 h[4]; uint2 u; } p;
    p.h[0] = (__bf16)v.x; p.h[1] = (__bf16)v.y;
    p.h[2] = (__bf16)v.z; p.h[3] = (__bf16)v.w;
    *(uint2*)(Xb + i) = p.u;
}

// W[z][R][C] f32 row-major  ->  WT[z][C][R] bf16 (n-major, contiguous K)
__global__ __launch_bounds__(256) void transpose_cvt(const float* __restrict__ src,
                                                     __bf16* __restrict__ dst,
                                                     int R, int C) {
    __shared__ __bf16 tile[32][33];
    const size_t eb = (size_t)blockIdx.z * R * C;
    const int r0 = blockIdx.x * 32;
    const int c0 = blockIdx.y * 32;
    const int tid = threadIdx.x;
    const int rr = tid >> 3;
    const int q4 = (tid & 7) * 4;

    float4 v = *(const float4*)(src + eb + (size_t)(r0 + rr) * C + c0 + q4);
    tile[q4 + 0][rr] = (__bf16)v.x;
    tile[q4 + 1][rr] = (__bf16)v.y;
    tile[q4 + 2][rr] = (__bf16)v.z;
    tile[q4 + 3][rr] = (__bf16)v.w;
    __syncthreads();

    union { __bf16 h[4]; uint2 u; } p;
    p.h[0] = tile[rr][q4 + 0];
    p.h[1] = tile[rr][q4 + 1];
    p.h[2] = tile[rr][q4 + 2];
    p.h[3] = tile[rr][q4 + 3];
    *(uint2*)(dst + eb + (size_t)(c0 + rr) * R + r0 + q4) = p.u;
}

// ---------------------------------------------------------------------------
// GEMM1: h = silu(X*Wg) * (X*Wu)
// block = 256 (8 waves). Block tile: 64 rows x 64 h-cols.
// Waves 0-3: gate 64x64 (32x32 each). Waves 4-7: up 64x64.
// Double-buffered async-LDS staging, 1 barrier per 32-K step, 4 WMMA/wave/step.
// ---------------------------------------------------------------------------
__global__ __launch_bounds__(256) void moe_gemm1(
    const __bf16* __restrict__ Xb, const __bf16* __restrict__ WguT,
    const int* __restrict__ offP, const int* __restrict__ tokl,
    __bf16* __restrict__ hbuf)
{
    __shared__ __bf16 As[2][64][32];   // [buf][m][k]  8 KB
    __shared__ __bf16 Bg[2][64][32];   // [buf][n][k]  8 KB
    __shared__ __bf16 Bu[2][64][32];   //              8 KB
    __shared__ float  gxch[64][65];    // gate exchange (epilogue) 16.6 KB
    __shared__ int    tks[64];

    const int row0  = blockIdx.x * 64;
    const int total = offP[EE];
    if (row0 >= total) return;
    int e = 0;
    #pragma unroll
    for (int q = 0; q < EE; ++q) if (row0 >= offP[q + 1]) e = q + 1;

    const int n0   = blockIdx.y * 64;
    const int tid  = threadIdx.x;
    const int lane = tid & 31;
    const int wave = tid >> 5;
    const int mat  = wave >> 2;                 // 0 = gate, 1 = up
    const int pos  = wave & 3;
    const int wm32 = (pos >> 1) * 32;
    const int wn32 = (pos & 1) * 32;
    const int kh   = lane >> 4;
    const int lq   = lane & 15;

    if (tid < 64) tks[tid] = tokl[row0 + tid];
    {   // zero both A buffers once (pad rows stay zero)
        uint4 z = make_uint4(0u, 0u, 0u, 0u);
        uint4* p = (uint4*)&As[0][0][0];
        p[tid] = z; p[tid + 256] = z;
    }
    __syncthreads();

    // staging: each thread owns one 16B chunk per matrix
    const int srow  = tid >> 2;                 // 0..63
    const int skoff = (tid & 3) * 8;            // element offset of 16B chunk
    const int ttok  = tks[srow];
    const __bf16* xsrc = (ttok >= 0) ? (Xb + (size_t)ttok * HH + skoff) : nullptr;
    const __bf16* gsrc = WguT + ((size_t)e * (2 * II) + n0 + srow) * HH + skoff;
    const __bf16* usrc = gsrc + (size_t)II * HH;

    auto stage = [&](int b, int kb) {
        if (xsrc) async_b128(&As[b][srow][skoff], xsrc + kb);
        async_b128(&Bg[b][srow][skoff], gsrc + kb);
        async_b128(&Bu[b][srow][skoff], usrc + kb);
    };

    v8f acc[2][2] = {{{}, {}}, {{}, {}}};

    stage(0, 0);
    int cur = 0;
    for (int kb = 0; kb < HH; kb += 32) {
        wait_async0();          // current buffer (loads overlapped prev compute)
        __syncthreads();        // visible to all waves; prev reads of other buf done
        if (kb + 32 < HH) stage(cur ^ 1, kb + 32);

        const __bf16 (*Bm)[32] = mat ? Bu[cur] : Bg[cur];
        BF16x16 a0, a1, b0, b1;
        a0.h[0] = *(const v8bf*)&As[cur][wm32 + lq][kh * 8];
        a0.h[1] = *(const v8bf*)&As[cur][wm32 + lq][16 + kh * 8];
        a1.h[0] = *(const v8bf*)&As[cur][wm32 + 16 + lq][kh * 8];
        a1.h[1] = *(const v8bf*)&As[cur][wm32 + 16 + lq][16 + kh * 8];
        b0.h[0] = *(const v8bf*)&Bm[wn32 + lq][kh * 8];
        b0.h[1] = *(const v8bf*)&Bm[wn32 + lq][16 + kh * 8];
        b1.h[0] = *(const v8bf*)&Bm[wn32 + 16 + lq][kh * 8];
        b1.h[1] = *(const v8bf*)&Bm[wn32 + 16 + lq][16 + kh * 8];

        acc[0][0] = wmma_bf16(a0, b0, acc[0][0]);
        acc[0][1] = wmma_bf16(a0, b1, acc[0][1]);
        acc[1][0] = wmma_bf16(a1, b0, acc[1][0]);
        acc[1][1] = wmma_bf16(a1, b1, acc[1][1]);
        cur ^= 1;
    }

    // Epilogue: gate waves publish f32 tile; up waves fuse silu(g)*u -> bf16
    if (mat == 0) {
        #pragma unroll
        for (int i = 0; i < 2; ++i)
            #pragma unroll
            for (int j = 0; j < 2; ++j)
                #pragma unroll
                for (int r = 0; r < 8; ++r)
                    gxch[wm32 + i * 16 + r + 8 * kh][wn32 + j * 16 + lq] = acc[i][j][r];
    }
    __syncthreads();
    if (mat == 1) {
        #pragma unroll
        for (int i = 0; i < 2; ++i) {
            #pragma unroll
            for (int j = 0; j < 2; ++j) {
                #pragma unroll
                for (int r = 0; r < 8; ++r) {
                    int row = wm32 + i * 16 + r + 8 * kh;
                    int col = wn32 + j * 16 + lq;
                    float g = gxch[row][col];
                    float u = acc[i][j][r];
                    float sig = __builtin_amdgcn_rcpf(1.0f + __expf(-g));
                    hbuf[(size_t)(row0 + row) * II + n0 + col] = (__bf16)(g * sig * u);
                }
            }
        }
    }
}

// ---------------------------------------------------------------------------
// GEMM2: out[t] += w * (h * Wd)
// block = 256 (8 waves, 2x4 grid of 32x32 tiles). Block tile: 64 rows x 128 cols.
// ---------------------------------------------------------------------------
__global__ __launch_bounds__(256) void moe_gemm2(
    const __bf16* __restrict__ WdT, const __bf16* __restrict__ hbuf,
    const int* __restrict__ offP, const int* __restrict__ tokl,
    const float* __restrict__ wgtl, float* __restrict__ out)
{
    __shared__ __bf16 As[2][64][32];    // 8 KB
    __shared__ __bf16 Bs[2][128][32];   // 16 KB
    __shared__ int    tks[64];
    __shared__ float  wls[64];

    const int row0  = blockIdx.x * 64;
    const int total = offP[EE];
    if (row0 >= total) return;
    int e = 0;
    #pragma unroll
    for (int q = 0; q < EE; ++q) if (row0 >= offP[q + 1]) e = q + 1;

    const int n0   = blockIdx.y * 128;
    const int tid  = threadIdx.x;
    const int lane = tid & 31;
    const int wave = tid >> 5;
    const int wm32 = (wave >> 2) * 32;          // 0 | 32
    const int wn32 = (wave & 3) * 32;           // 0..96
    const int kh   = lane >> 4;
    const int lq   = lane & 15;

    if (tid < 64) { tks[tid] = tokl[row0 + tid]; wls[tid] = wgtl[row0 + tid]; }
    __syncthreads();

    const int srow  = tid >> 2;
    const int skoff = (tid & 3) * 8;
    const __bf16* hsrc = hbuf + (size_t)(row0 + srow) * II + skoff;  // pad rows = 0
    // B: 128 n-rows -> two chunks per thread
    const int bn1   = (tid + 256) >> 2;
    const __bf16* dsrc0 = WdT + ((size_t)e * HH + n0 + srow) * II + skoff;
    const __bf16* dsrc1 = WdT + ((size_t)e * HH + n0 + bn1) * II + ((tid + 256) & 3) * 8;
    const int bk1 = ((tid + 256) & 3) * 8;

    auto stage = [&](int b, int kb) {
        async_b128(&As[b][srow][skoff], hsrc + kb);
        async_b128(&Bs[b][srow][skoff], dsrc0 + kb);
        async_b128(&Bs[b][bn1][bk1],    dsrc1 + kb);
    };

    v8f acc[2][2] = {{{}, {}}, {{}, {}}};

    stage(0, 0);
    int cur = 0;
    for (int kb = 0; kb < II; kb += 32) {
        wait_async0();
        __syncthreads();
        if (kb + 32 < II) stage(cur ^ 1, kb + 32);

        BF16x16 a0, a1, b0, b1;
        a0.h[0] = *(const v8bf*)&As[cur][wm32 + lq][kh * 8];
        a0.h[1] = *(const v8bf*)&As[cur][wm32 + lq][16 + kh * 8];
        a1.h[0] = *(const v8bf*)&As[cur][wm32 + 16 + lq][kh * 8];
        a1.h[1] = *(const v8bf*)&As[cur][wm32 + 16 + lq][16 + kh * 8];
        b0.h[0] = *(const v8bf*)&Bs[cur][wn32 + lq][kh * 8];
        b0.h[1] = *(const v8bf*)&Bs[cur][wn32 + lq][16 + kh * 8];
        b1.h[0] = *(const v8bf*)&Bs[cur][wn32 + 16 + lq][kh * 8];
        b1.h[1] = *(const v8bf*)&Bs[cur][wn32 + 16 + lq][16 + kh * 8];

        acc[0][0] = wmma_bf16(a0, b0, acc[0][0]);
        acc[0][1] = wmma_bf16(a0, b1, acc[0][1]);
        acc[1][0] = wmma_bf16(a1, b0, acc[1][0]);
        acc[1][1] = wmma_bf16(a1, b1, acc[1][1]);
        cur ^= 1;
    }

    #pragma unroll
    for (int i = 0; i < 2; ++i) {
        #pragma unroll
        for (int j = 0; j < 2; ++j) {
            #pragma unroll
            for (int r = 0; r < 8; ++r) {
                int row = wm32 + i * 16 + r + 8 * kh;
                int t   = tks[row];
                if (t >= 0) {
                    atomicAdd(out + (size_t)t * HH + n0 + wn32 + j * 16 + lq,
                              acc[i][j][r] * wls[row]);
                }
            }
        }
    }
}

// ---------------------------------------------------------------------------
// Launch
// ---------------------------------------------------------------------------
extern "C" void kernel_launch(void* const* d_in, const int* in_sizes, int n_in,
                              void* d_out, int out_size, void* d_ws, size_t ws_size,
                              hipStream_t stream) {
    const float* X    = (const float*)d_in[0];   // hidden_states  (T,H)
    const float* aff  = (const float*)d_in[1];   // expert_affinities (T,E)
    const float* Wgu  = (const float*)d_in[2];   // gate_up_proj (E,H,2I)
    const float* Wd   = (const float*)d_in[3];   // down_proj    (E,I,H)
    const int*   eidx = (const int*)d_in[4];     // expert_index (T,K)
    float*       out  = (float*)d_out;           // (T,H) fp32

    char* ws = (char*)d_ws;
    int*    counts = (int*)(ws + WS_COUNTS);
    int*    fill   = counts + 8;
    int*    offP   = (int*)(ws + WS_OFFP);
    int*    tokl   = (int*)(ws + WS_TOK);
    float*  wgtl   = (float*)(ws + WS_WGT);
    __bf16* Xb     = (__bf16*)(ws + WS_XB);
    __bf16* hbuf   = (__bf16*)(ws + WS_HBUF);
    __bf16* WguT   = (__bf16*)(ws + WS_WGUT);
    __bf16* WdT    = (__bf16*)(ws + WS_WDT);

    // routing
    moe_init<<<(TT * HH + 255) / 256, 256, 0, stream>>>(out, counts, tokl);
    moe_count<<<(TT * KK + 255) / 256, 256, 0, stream>>>(eidx, counts);
    moe_scan<<<1, 32, 0, stream>>>(counts, offP);
    moe_scatter<<<(TT * KK + 255) / 256, 256, 0, stream>>>(eidx, aff, offP, fill, tokl, wgtl);

    // one-time bf16 conversion + weight transposition (n-major, contiguous K)
    cvt_x<<<(TT * HH / 4 + 255) / 256, 256, 0, stream>>>(X, Xb);
    dim3 tg1(HH / 32, (2 * II) / 32, EE);
    transpose_cvt<<<tg1, 256, 0, stream>>>(Wgu, WguT, HH, 2 * II);
    dim3 tg2(II / 32, HH / 32, EE);
    transpose_cvt<<<tg2, 256, 0, stream>>>(Wd, WdT, II, HH);

    // routed grouped GEMMs
    dim3 g1(SLOT_CAP / 64, II / 64);    // 136 x 22
    moe_gemm1<<<g1, 256, 0, stream>>>(Xb, WguT, offP, tokl, hbuf);
    dim3 g2(SLOT_CAP / 64, HH / 128);   // 136 x 16
    moe_gemm2<<<g2, 256, 0, stream>>>(WdT, hbuf, offP, tokl, wgtl, out);
}
